// YOLOv2_78082505441478
// MI455X (gfx1250) — matrix operand order repaired and checked by reading, not compiled, not run on hardware
//
#include <hip/hip_runtime.h>
#include <math.h>

// ---------------------------------------------------------------------------
// YOLOv2 head on gfx1250 (MI455X).
//   K1: TDM tile load -> LDS, f16 transpose, per-wave f16 weight panels,
//       v_wmma_f32_16x16x32_f16 GEMM (K=512), fused sigmoid/score,
//       exact per-block top-100.
//   K2: merge 512 candidate lists -> global sorted top-100 + box decode.
// Memory-bound: 67MB feature reads @ 23.3TB/s ~= 2.9us floor; 0.87MB weights
// are L2-resident (192MB L2). Inner loop: ds_load_b128 x6 + WMMA x2 per chunk.
// ---------------------------------------------------------------------------

typedef _Float16     v16h  __attribute__((ext_vector_type(16)));
typedef _Float16     v8h   __attribute__((ext_vector_type(8)));
typedef _Float16     v4h   __attribute__((ext_vector_type(4)));
typedef _Float16     v2h   __attribute__((ext_vector_type(2)));
typedef float        v8f   __attribute__((ext_vector_type(8)));
typedef unsigned int u32x4 __attribute__((ext_vector_type(4)));
typedef int          i32x4 __attribute__((ext_vector_type(4)));
typedef int          i32x8 __attribute__((ext_vector_type(8)));

#define HW_TOT   16384            // 128*128 spatial positions
#define KDIM     512              // head dim (GEMM K)
#define NT       32               // spatial positions per block (GEMM N tile)
#define NBLK     (HW_TOT / NT)    // 512 blocks
#define NSCORE   (NT * 400)       // 12800 scores per block
#define NCAND    (NBLK * 100)     // 51200 global candidates
#define TOPK     100
#define BSTRIDE  520              // padded f16 row stride (bank-conflict break)

// Kernel-1 dynamic LDS layout (bytes):
//   [0,      33280)  btC: cls features f16 transposed [32 n][520 k]
//   [33280,  66560)  btR: reg features f16 transposed [32 n][520 k]
//   [66560, 199680)  R2: f32 staging tiles (2x 512x32) -> per-wave A panels
//                        (8 x 16x520 f16) -> score array (12800 f32)
//   [199680,254976)  outL: logit rows f32 [432][32]
//   [254976,257024)  argmax reduce val[256] + idx[256]
//   [257024,258752)  biasL: unified bias row f32 [432]
#define OFF_BTC   0
#define OFF_BTR   33280
#define OFF_R2    66560
#define OFF_T32C  66560
#define OFF_T32R  132096
#define APANEL_SZ (16 * BSTRIDE * 2)     // 16640 bytes per wave
#define OFF_OUTL  199680
#define OFF_REDV  254976
#define OFF_REDI  256000
#define OFF_BIAS  257024
#define LDS1_TOTAL 258752

__device__ __constant__ float c_anchor_w[5] = {38.08f, 89.28f, 144.96f, 257.92f, 330.24f};
__device__ __constant__ float c_anchor_h[5] = {63.36f, 146.88f, 285.44f, 169.28f, 340.80f};

__device__ __forceinline__ float sigf(float x) { return 1.0f / (1.0f + __expf(-x)); }

#if defined(__has_builtin)
#if __has_builtin(__builtin_amdgcn_tensor_load_to_lds)
#define HAVE_TDM 1
#endif
#endif
#ifndef HAVE_TDM
#define HAVE_TDM 0
#endif

#if HAVE_TDM
// 2D TDM tile load: [KDIM rows x NT cols] f32, row stride HW_TOT elements,
// packed contiguously into LDS at lds_byte_off. Wave-uniform; EXEC ignored.
__device__ __forceinline__ void tdm_load_tile_512x32(const float* gtile,
                                                     unsigned lds_byte_off) {
  unsigned long long ga = (unsigned long long)(size_t)(const void*)gtile;
  u32x4 g0;
  g0[0] = 1u;                                            // D# count=1 (valid, user)
  g0[1] = lds_byte_off;                                  // lds_addr (bytes)
  g0[2] = (unsigned)ga;                                  // global_addr[31:0]
  g0[3] = (unsigned)((ga >> 32) & 0x01FFFFFFull) | (2u << 30); // addr[56:32]|type=2
  i32x8 g1;
  g1[0] = (int)(2u << 16);                               // data_size=4B
  g1[1] = (int)(((unsigned)HW_TOT & 0xFFFFu) << 16);     // tensor_dim0 lo16
  g1[2] = (int)((((unsigned)HW_TOT >> 16) & 0xFFFFu) |
                (((unsigned)KDIM & 0xFFFFu) << 16));     // tdim0 hi | tdim1 lo
  g1[3] = (int)((((unsigned)KDIM >> 16) & 0xFFFFu) |
                ((unsigned)NT << 16));                   // tdim1 hi | tile_dim0=32
  g1[4] = (int)(unsigned)KDIM;                           // tile_dim1=512
  g1[5] = (int)(unsigned)HW_TOT;                         // tensor_dim0_stride lo32
  g1[6] = 0;
  g1[7] = 0;
  i32x4 g2 = {0, 0, 0, 0};
  i32x4 g3 = {0, 0, 0, 0};
  i32x8 g4 = {0, 0, 0, 0, 0, 0, 0, 0};
  __builtin_amdgcn_tensor_load_to_lds(g0, g1, g2, g3, g4, 0);
}
#endif

// ---------------------------------------------------------------------------
// Kernel 1
// ---------------------------------------------------------------------------
__global__ __launch_bounds__(256) void yolo_head_topk(
    const float* __restrict__ cls_feat, const float* __restrict__ reg_feat,
    const float* __restrict__ w_obj, const float* __restrict__ b_obj,
    const float* __restrict__ w_cls, const float* __restrict__ b_cls,
    const float* __restrict__ w_reg, const float* __restrict__ b_reg,
    float* __restrict__ reg_ws, float* __restrict__ cand_s,
    int* __restrict__ cand_i) {
  extern __shared__ char smem[];
  _Float16* btC    = (_Float16*)(smem + OFF_BTC);
  _Float16* btR    = (_Float16*)(smem + OFF_BTR);
  float*    t32C   = (float*)(smem + OFF_T32C);
  float*    t32R   = (float*)(smem + OFF_T32R);
  float*    outL   = (float*)(smem + OFF_OUTL);
  float*    redV   = (float*)(smem + OFF_REDV);
  int*      redI   = (int*)  (smem + OFF_REDI);
  float*    biasL  = (float*)(smem + OFF_BIAS);
  float*    scores = (float*)(smem + OFF_R2);  // overlays panels after GEMM

  const int tid   = threadIdx.x;
  const int lane  = tid & 31;
  const int wave  = tid >> 5;
  const int nbase = blockIdx.x * NT;

  if (tid == 0) {
    __builtin_prefetch(w_cls, 0, 1);
    __builtin_prefetch(w_reg, 0, 1);
  }

  // ---- Stage both feature tiles (f32) into LDS via the Tensor Data Mover ----
#if HAVE_TDM
  if (wave == 0) {
    tdm_load_tile_512x32(cls_feat + nbase, (unsigned)(size_t)(void*)t32C);
    tdm_load_tile_512x32(reg_feat + nbase, (unsigned)(size_t)(void*)t32R);
    __builtin_amdgcn_s_wait_tensorcnt(0);
  }
#else
  for (int i = tid; i < KDIM * (NT / 4); i += 256) {
    int r = i >> 3, c = i & 7;
    ((float4*)t32C)[(r << 3) + c] =
        *(const float4*)(cls_feat + (size_t)r * HW_TOT + nbase + c * 4);
    ((float4*)t32R)[(r << 3) + c] =
        *(const float4*)(reg_feat + (size_t)r * HW_TOT + nbase + c * 4);
  }
#endif

  // ---- Unified bias row (once per block): [0,400)=cls, [400,405)=obj,
  //      [405,425)=reg, rest 0. Removes all bias branching from the GEMM. ----
  for (int i = tid; i < 432; i += 256) {
    float b = 0.0f;
    if (i < 400)      b = b_cls[i];
    else if (i < 405) b = b_obj[i - 400];
    else if (i < 425) b = b_reg[i - 405];
    biasL[i] = b;
  }
  __syncthreads();

  // ---- Transpose + f16 convert: bt[n][k], padded stride 520 ----
  for (int i = tid; i < KDIM * NT; i += 256) {
    const int k = i >> 5, n = i & 31;        // consecutive tid -> consecutive n
    btC[n * BSTRIDE + k] = (_Float16)t32C[k * NT + n];
    btR[n * BSTRIDE + k] = (_Float16)t32R[k * NT + n];
  }
  __syncthreads();

  // ---- WMMA GEMM: 27 M-tiles (25 cls + 2 obj/reg) x 2 N-subtiles, K=512 ----
  // A (16x32 f16): lane holds M=lane&15, K runs {kk+ksA..+7} and {kk+16+ksA..+7}
  // B (32x16 f16): lane holds N=lane&15, K runs {kk+ksB..+7} and {kk+16+ksB..+7}
  const int ksA  = (lane >= 16) ? 8 : 0;
  const int ksB  = (lane >= 16) ? 8 : 0;
  const int mrow = lane & 15;
  _Float16* panel = (_Float16*)(smem + OFF_R2 + wave * APANEL_SZ);

  for (int mt = wave; mt < 27; mt += 8) {
    const bool      isCls = (mt < 25);
    const _Float16* bt    = isCls ? btC : btR;
    const int       mt2   = isCls ? mt : (mt - 25);

    // ---- Stage this wave's 16x512 f16 weight panel ----
    if (isCls) {
      // Branch-free: rows are contiguous in w_cls; (row,kp) linearizes to 4*i.
      const float* wbase = w_cls + (size_t)mt2 * (16 * KDIM);
      for (int i = lane; i < 16 * 128; i += 32) {
        const float4 wv = *(const float4*)(wbase + 4 * i);
        v4h p;
        p[0] = (_Float16)wv.x; p[1] = (_Float16)wv.y;
        p[2] = (_Float16)wv.z; p[3] = (_Float16)wv.w;
        // panel half-offset: row*520 + 4*kp = 4*i + 8*(i>>7)
        *(v4h*)(panel + 4 * i + 8 * (i >> 7)) = p;
      }
    } else {
      // obj(5) + reg(20) + zero-pad rows; runs for only 2 of 27 m-tiles.
      for (int i = lane; i < 16 * 256; i += 32) {
        const int row = i >> 8;                 // 0..15
        const int kp  = i & 255;                // float2 index
        const int O   = mt2 * 16 + row;
        const float* rp = nullptr;
        if (O < 5)       rp = w_obj + (size_t)O * KDIM;
        else if (O < 25) rp = w_reg + (size_t)(O - 5) * KDIM;
        float2 wv = {0.0f, 0.0f};
        if (rp) wv = *(const float2*)(rp + 2 * kp);
        v2h p;
        p[0] = (_Float16)wv.x;
        p[1] = (_Float16)wv.y;
        *(v2h*)(panel + row * BSTRIDE + 2 * kp) = p;
      }
    }
    asm volatile("s_wait_dscnt 0x0" ::: "memory");  // panel visible to this wave

    const _Float16* pa  = panel + mrow * BSTRIDE + ksA;
    const _Float16* pb0 = bt + mrow * BSTRIDE + ksB;
    const _Float16* pb1 = bt + (16 + mrow) * BSTRIDE + ksB;

    v8f acc0 = {0.f, 0.f, 0.f, 0.f, 0.f, 0.f, 0.f, 0.f};
    v8f acc1 = acc0;
#pragma unroll 2
    for (int kk = 0; kk < KDIM; kk += 32) {
      const v8h alo = *(const v8h*)(pa + kk);
      const v8h ahi = *(const v8h*)(pa + kk + 16);
      const v8h b0l = *(const v8h*)(pb0 + kk);
      const v8h b0h = *(const v8h*)(pb0 + kk + 16);
      const v8h b1l = *(const v8h*)(pb1 + kk);
      const v8h b1h = *(const v8h*)(pb1 + kk + 16);
      v16h af, b0, b1;
#pragma unroll
      for (int h = 0; h < 8; ++h) {
        af[h] = alo[h]; af[h + 8] = ahi[h];
        b0[h] = b0l[h]; b0[h + 8] = b0h[h];
        b1[h] = b1l[h]; b1[h + 8] = b1h[h];
      }
      acc0 = __builtin_amdgcn_wmma_f32_16x16x32_f16(false, af, false, b0,
                                                    (short)0, acc0, false, false);
      acc1 = __builtin_amdgcn_wmma_f32_16x16x32_f16(false, af, false, b1,
                                                    (short)0, acc1, false, false);
    }

    // C/D layout: VGPR v holds M = v + (lane>=16 ? 8:0), N = lane&15.
    // Out-row = O (cls tile) or 400+O (obj/reg tile, valid iff O<25).
    const int Obase  = mt2 * 16 + ((lane >= 16) ? 8 : 0);
    const int rowAdd = isCls ? 0 : 400;
#pragma unroll
    for (int v = 0; v < 8; ++v) {
      const int O = Obase + v;
      if (isCls || O < 25) {
        const int   row  = rowAdd + O;
        const float bias = biasL[row];
        outL[row * NT + mrow]      = acc0[v] + bias;
        outL[row * NT + 16 + mrow] = acc1[v] + bias;
      }
    }
  }
  __syncthreads();

  // ---- Fused scores: sqrt(sigmoid(obj)*sigmoid(cls)); reg -> workspace ----
  for (int s = tid; s < NSCORE; s += 256) {
    const int n = s / 400;
    const int r = s - n * 400;          // = a*80 + c
    const int a = r / 80;
    const float o  = outL[(400 + a) * NT + n];
    const float cl = outL[r * NT + n];
    scores[s] = sqrtf(sigf(o) * sigf(cl));
  }
  for (int i = tid; i < 20 * NT; i += 256) {
    const int o = i >> 5, n = i & 31;
    reg_ws[o * HW_TOT + nbase + n] = outL[(405 + o) * NT + n];
  }
  __syncthreads();

  // ---- Exact block-local top-100 via argmax/invalidate ----
  for (int rk = 0; rk < TOPK; ++rk) {
    float bv = -1e30f;
    int bi = 0;
    for (int s = tid; s < NSCORE; s += 256)
      if (scores[s] > bv) { bv = scores[s]; bi = s; }
    redV[tid] = bv; redI[tid] = bi;
    __syncthreads();
    for (int st = 128; st > 0; st >>= 1) {
      if (tid < st && redV[tid + st] > redV[tid]) {
        redV[tid] = redV[tid + st]; redI[tid] = redI[tid + st];
      }
      __syncthreads();
    }
    if (tid == 0) {
      const int s = redI[0];
      cand_s[blockIdx.x * TOPK + rk] = redV[0];
      cand_i[blockIdx.x * TOPK + rk] = blockIdx.x * NSCORE + s;
      scores[s] = -1e30f;
    }
    __syncthreads();
  }
}

// ---------------------------------------------------------------------------
// Kernel 2: merge candidate lists to a global sorted top-100 + box decode.
// ---------------------------------------------------------------------------
__global__ __launch_bounds__(1024) void yolo_merge_decode(
    const float* __restrict__ reg_ws, const float* __restrict__ cand_s,
    const int* __restrict__ cand_i, float* __restrict__ out) {
  extern __shared__ char smem[];
  float* cs   = (float*)smem;                      // [51200]
  float* redV = (float*)(smem + NCAND * 4);        // [1024]
  int*   redI = (int*)  (smem + NCAND * 4 + 4096); // [1024]
  __shared__ float tks[TOPK];
  __shared__ int   tki[TOPK];
  const int tid = threadIdx.x;

  for (int i = tid; i < NCAND; i += 1024) cs[i] = cand_s[i];
  __syncthreads();

  for (int rk = 0; rk < TOPK; ++rk) {
    float bv = -1e30f;
    int bi = 0;
    for (int i = tid; i < NCAND; i += 1024)
      if (cs[i] > bv) { bv = cs[i]; bi = i; }
    redV[tid] = bv; redI[tid] = bi;
    __syncthreads();
    for (int st = 512; st > 0; st >>= 1) {
      if (tid < st && redV[tid + st] > redV[tid]) {
        redV[tid] = redV[tid + st]; redI[tid] = redI[tid + st];
      }
      __syncthreads();
    }
    if (tid == 0) { tks[rk] = redV[0]; tki[rk] = redI[0]; cs[redI[0]] = -1e30f; }
    __syncthreads();
  }

  if (tid < TOPK) {
    const float sc   = tks[tid];
    const int   flat = cand_i[tki[tid]];
    const bool  keep = sc > 0.01f;
    const int label = flat % 80;
    const int aidx  = flat / 80;
    const int a     = aidx % 5;
    const int n     = aidx / 5;        // spatial index = h*128 + w
    const float gx = (float)(n & 127);
    const float gy = (float)(n >> 7);
    const float t0 = reg_ws[(a * 4 + 0) * HW_TOT + n];
    const float t1 = reg_ws[(a * 4 + 1) * HW_TOT + n];
    const float t2 = reg_ws[(a * 4 + 2) * HW_TOT + n];
    const float t3 = reg_ws[(a * 4 + 3) * HW_TOT + n];
    const float cx = (sigf(t0) + gx) * 32.0f;
    const float cy = (sigf(t1) + gy) * 32.0f;
    const float bw = __expf(t2) * c_anchor_w[a];
    const float bh = __expf(t3) * c_anchor_h[a];
    float x0 = cx - 0.5f * bw, y0 = cy - 0.5f * bh;
    float x1 = cx + 0.5f * bw, y1 = cy + 0.5f * bh;
    if (!keep) { x0 = y0 = x1 = y1 = 0.0f; }
    out[tid * 4 + 0] = x0;
    out[tid * 4 + 1] = y0;
    out[tid * 4 + 2] = x1;
    out[tid * 4 + 3] = y1;
    out[400 + tid] = keep ? sc : 0.0f;
    out[500 + tid] = keep ? (float)label : -1.0f;
  }
}

extern "C" void kernel_launch(void* const* d_in, const int* in_sizes, int n_in,
                              void* d_out, int out_size, void* d_ws, size_t ws_size,
                              hipStream_t stream) {
  (void)in_sizes; (void)n_in; (void)out_size; (void)ws_size;
  const float* cls_feat = (const float*)d_in[0];
  const float* reg_feat = (const float*)d_in[1];
  const float* w_obj = (const float*)d_in[2];
  const float* b_obj = (const float*)d_in[3];
  const float* w_cls = (const float*)d_in[4];
  const float* b_cls = (const float*)d_in[5];
  const float* w_reg = (const float*)d_in[6];
  const float* b_reg = (const float*)d_in[7];

  // Workspace: [reg logits 20x16384 f32 | candidate scores | candidate indices]
  float* reg_ws = (float*)d_ws;
  float* cand_s = (float*)((char*)d_ws + (size_t)20 * HW_TOT * 4);
  int*   cand_i = (int*)((char*)d_ws + (size_t)20 * HW_TOT * 4 + (size_t)NCAND * 4);

  yolo_head_topk<<<NBLK, 256, LDS1_TOTAL, stream>>>(
      cls_feat, reg_feat, w_obj, b_obj, w_cls, b_cls, w_reg, b_reg,
      reg_ws, cand_s, cand_i);
  yolo_merge_decode<<<1, 1024, NCAND * 4 + 8192, stream>>>(
      reg_ws, cand_s, cand_i, (float*)d_out);
}
